// Renet3d_59837484368065
// MI455X (gfx1250) — matrix operand order repaired and checked by reading, not compiled, not run on hardware
//
#include <hip/hip_runtime.h>
#include <hip/hip_bf16.h>

// MI455X / gfx1250, wave32. All matrix math via v_wmma_f32_16x16x32_bf16.

typedef __attribute__((ext_vector_type(16))) __bf16 v16bf;
typedef __attribute__((ext_vector_type(8)))  __bf16 v8bf;
typedef __attribute__((ext_vector_type(8)))  float  v8f;

__device__ __forceinline__ float sigm_(float x) { return 1.0f / (1.0f + __expf(-x)); }

// ---------------- fp32 -> bf16 convert ----------------
__global__ void cvt_f32_bf16(const float* __restrict__ in, __bf16* __restrict__ out, int n) {
  int i = blockIdx.x * blockDim.x + threadIdx.x;
  if (i < n) out[i] = (__bf16)in[i];
}

// ---------------- generic GEMM: Out[m,n] = sum_k A(m,k) * W[n,k] + bias[n] ----------------
// Row index m (< 8192) decomposes into nibble fields f3=m>>12, f2=(m>>8)&15,
// f1=(m>>4)&15, f0=m&15 so arbitrary (B,C,D,H,W) permutations are stride sets.
// A_F32: A fp32, element addr = base + k*sk.  else: A bf16, contiguous in k (sk==1).
template <bool A_F32, bool OUT_F32>
__global__ __launch_bounds__(256) void gemm_bias_wmma(
    const void* __restrict__ Aptr,
    long s3, long s2, long s1, long s0, long sk,
    const __bf16* __restrict__ W, int K,
    const float* __restrict__ bias,
    void* __restrict__ Optr,
    long o3, long o2, long o1, long o0, long ok,
    int nTilesM, int nTilesN)
{
  const int lane = threadIdx.x & 31;
  const int wave = threadIdx.x >> 5;
  int tile = blockIdx.x * (blockDim.x >> 5) + wave;
  if (tile >= nTilesM * nTilesN) return;
  const int tM = tile / nTilesN;
  const int tN = tile - tM * nTilesN;
  const int hi = lane >> 4;       // 0 or 1
  const int lo = lane & 15;

  // A fragment row for this lane (16-bit A 16x32 layout: lane&15 = M row)
  const int rowA = tM * 16 + lo;
  const long abase = (long)(rowA >> 12) * s3 + (long)((rowA >> 8) & 15) * s2 +
                     (long)((rowA >> 4) & 15) * s1 + (long)(rowA & 15) * s0;
  // B fragment column (lane&15 = N col; this code's k order: 16 contiguous K per lane half, same as A)
  const int colB = tN * 16 + lo;
  const __bf16* wrow = W + (long)colB * K;

  v8f acc = {};
  for (int kk = 0; kk < K; kk += 32) {
    v16bf a;
    const int kb = kk + hi * 16;                // elems 0..15 -> K=kb..kb+15
    if constexpr (A_F32) {
      const float* Af = (const float*)Aptr;
#pragma unroll
      for (int e = 0; e < 8; ++e) a[e] = (__bf16)Af[abase + (long)(kb + e) * sk];
#pragma unroll
      for (int e = 0; e < 8; ++e) a[8 + e] = (__bf16)Af[abase + (long)(kb + 8 + e) * sk];
    } else {
      const __bf16* Ab = (const __bf16*)Aptr;
      v8bf a0 = *(const v8bf*)(Ab + abase + kb);
      v8bf a1 = *(const v8bf*)(Ab + abase + kb + 8);
#pragma unroll
      for (int e = 0; e < 8; ++e) { a[e] = a0[e]; a[8 + e] = a1[e]; }
    }
    v16bf b = *(const v16bf*)(wrow + kk + hi * 16);
    acc = __builtin_amdgcn_wmma_f32_16x16x32_bf16(false, a, false, b, (short)0, acc,
                                                  false, false);
  }

  const float bv = bias[colB];
#pragma unroll
  for (int j = 0; j < 8; ++j) {                 // C/D layout: VGPR j <-> M = j + 8*hi
    const int M = tM * 16 + j + hi * 8;
    const long oadr = (long)(M >> 12) * o3 + (long)((M >> 8) & 15) * o2 +
                      (long)((M >> 4) & 15) * o1 + (long)(M & 15) * o0 + (long)colB * ok;
    const float v = acc[j] + bv;
    if constexpr (OUT_F32) ((float*)Optr)[oadr] = v;
    else                   ((__bf16*)Optr)[oadr] = (__bf16)v;
  }
}

// ---------------- persistent LSTM scan (one direction per blockIdx.y) ----------------
// XW: t-major [t(16)][gate_col(1024)][r(512)] bf16 = x@Wih^T+b, so a lane's 8
//     C/D fragment elements (consecutive r) are one contiguous v8bf load.
// Whh: [1024][256] bf16.  Out: [512][16][512] bf16, fwd half at +0, bwd at +256.
// Block = 256 threads (8 waves) owns 16 sequences; h (bf16) + c (fp32) live in LDS.
// The K loop start rotates with the timestep so LICM cannot hoist the 64 Whh
// B-fragments (512 VGPRs!) out of the step loop -- they must stream from L2.
__global__ __launch_bounds__(256) void lstm_scan_wmma(
    const __bf16* __restrict__ XW_f, const __bf16* __restrict__ XW_b,
    const __bf16* __restrict__ Whh_f, const __bf16* __restrict__ Whh_b,
    __bf16* __restrict__ Out)
{
  __shared__ __bf16 h_lds[16][256];
  __shared__ float  c_lds[16][256];

  const int lane = threadIdx.x & 31;
  const int wv   = threadIdx.x >> 5;            // 0..7 -> hidden cols [32*wv, 32*wv+32)
  const int hi = lane >> 4, lo = lane & 15;
  const int dir = blockIdx.y;
  const __bf16* XW  = dir ? XW_b  : XW_f;
  const __bf16* Whh = dir ? Whh_b : Whh_f;
  const int r0 = blockIdx.x * 16;

  for (int i = threadIdx.x; i < 16 * 256; i += blockDim.x) {
    (&h_lds[0][0])[i] = (__bf16)0.0f;
    (&c_lds[0][0])[i] = 0.0f;
  }
  __syncthreads();

#pragma clang loop unroll(disable)
  for (int step = 0; step < 16; ++step) {
    const int t = dir ? (15 - step) : step;

    // init accumulators from the input projection: one b128 load per (gate, j)
    v8f acc[4][2];
#pragma unroll
    for (int q = 0; q < 4; ++q)
#pragma unroll
      for (int j = 0; j < 2; ++j) {
        const int col = q * 256 + wv * 32 + j * 16 + lo;
        const v8bf xv = *(const v8bf*)(XW + ((long)t * 1024 + col) * 512 + r0 + hi * 8);
#pragma unroll
        for (int e = 0; e < 8; ++e) acc[q][j][e] = (float)xv[e];
      }

    // gates += h @ Whh^T   (K = 256, 64 WMMAs per wave per step).
    // kk start rotates with step: reduction order changes, result identical
    // (modulo fp rounding), but Whh loads become step-variant -> no hoisting.
#pragma unroll
    for (int ki = 0; ki < 8; ++ki) {
      const int kk = ((ki + step) & 7) * 32;
      v16bf a;
      const int kb = kk + hi * 16;
#pragma unroll
      for (int e = 0; e < 8; ++e) { a[e] = h_lds[lo][kb + e]; a[8 + e] = h_lds[lo][kb + 8 + e]; }
#pragma unroll
      for (int q = 0; q < 4; ++q)
#pragma unroll
        for (int j = 0; j < 2; ++j) {
          const int col = q * 256 + wv * 32 + j * 16 + lo;
          v16bf b = *(const v16bf*)(Whh + (long)col * 256 + kk + hi * 16);
          acc[q][j] = __builtin_amdgcn_wmma_f32_16x16x32_bf16(false, a, false, b, (short)0,
                                                              acc[q][j], false, false);
        }
    }
    __syncthreads();   // everyone done reading old h

    // elementwise cell update (PyTorch gate order i,f,g,o)
#pragma unroll
    for (int j = 0; j < 2; ++j)
#pragma unroll
      for (int e = 0; e < 8; ++e) {
        const int m  = e + hi * 8;
        const int hc = wv * 32 + j * 16 + lo;
        const float iv = acc[0][j][e], fv = acc[1][j][e];
        const float gv = acc[2][j][e], ov = acc[3][j][e];
        const float cn = sigm_(fv) * c_lds[m][hc] + sigm_(iv) * tanhf(gv);
        const float hv = sigm_(ov) * tanhf(cn);
        c_lds[m][hc] = cn;
        h_lds[m][hc] = (__bf16)hv;
        Out[((long)(r0 + m) * 16 + t) * 512 + dir * 256 + hc] = (__bf16)hv;
      }
    __syncthreads();   // new h visible before next step's reads
  }
}

// ---------------- host orchestration ----------------
extern "C" void kernel_launch(void* const* d_in, const int* in_sizes, int n_in,
                              void* d_out, int out_size, void* d_ws, size_t ws_size,
                              hipStream_t stream) {
  (void)in_sizes; (void)n_in; (void)out_size; (void)ws_size;
  const float* x   = (const float*)d_in[0];
  const float* vwf = (const float*)d_in[1];  const float* vuf = (const float*)d_in[2];
  const float* vbf = (const float*)d_in[3];  const float* vwb = (const float*)d_in[4];
  const float* vub = (const float*)d_in[5];  const float* vbb = (const float*)d_in[6];
  const float* hwf = (const float*)d_in[7];  const float* huf = (const float*)d_in[8];
  const float* hbf = (const float*)d_in[9];  const float* hwb = (const float*)d_in[10];
  const float* hub = (const float*)d_in[11]; const float* hbb = (const float*)d_in[12];
  const float* c2w = (const float*)d_in[13]; const float* c2b = (const float*)d_in[14];
  const float* c3w = (const float*)d_in[15]; const float* c3b = (const float*)d_in[16];

  char* ws = (char*)d_ws; size_t off = 0;
  auto alloc = [&](size_t nbytes) -> char* {
    char* p = ws + off; off = (off + nbytes + 255) & ~(size_t)255; return p;
  };
  __bf16* vwf_bf = (__bf16*)alloc(1024 * 64  * 2);
  __bf16* vuf_bf = (__bf16*)alloc(1024 * 256 * 2);
  __bf16* vwb_bf = (__bf16*)alloc(1024 * 64  * 2);
  __bf16* vub_bf = (__bf16*)alloc(1024 * 256 * 2);
  __bf16* hwf_bf = (__bf16*)alloc(1024 * 512 * 2);
  __bf16* huf_bf = (__bf16*)alloc(1024 * 256 * 2);
  __bf16* hwb_bf = (__bf16*)alloc(1024 * 512 * 2);
  __bf16* hub_bf = (__bf16*)alloc(1024 * 256 * 2);
  __bf16* c2w_bf = (__bf16*)alloc(64 * 512 * 2);
  __bf16* c3w_bf = (__bf16*)alloc(64 * 64  * 2);
  __bf16* XWf  = (__bf16*)alloc((size_t)8192 * 1024 * 2);   // t-major frag layout
  __bf16* XWb  = (__bf16*)alloc((size_t)8192 * 1024 * 2);
  __bf16* vout = (__bf16*)alloc((size_t)512 * 16 * 512 * 2);  // [(n*16+h)*16+w][512]
  __bf16* hout = (__bf16*)alloc((size_t)512 * 16 * 512 * 2);  // [(n*16+w)*16+h][512]
  float*  x5a  = (float*)alloc((size_t)524288 * 4);
  float*  x5b  = (float*)alloc((size_t)524288 * 4);

  auto cvt = [&](const float* in, __bf16* out, int n) {
    cvt_f32_bf16<<<(n + 255) / 256, 256, 0, stream>>>(in, out, n);
  };
  cvt(vwf, vwf_bf, 1024 * 64);  cvt(vuf, vuf_bf, 1024 * 256);
  cvt(vwb, vwb_bf, 1024 * 64);  cvt(vub, vub_bf, 1024 * 256);
  cvt(hwf, hwf_bf, 1024 * 512); cvt(huf, huf_bf, 1024 * 256);
  cvt(hwb, hwb_bf, 1024 * 512); cvt(hub, hub_bf, 1024 * 256);
  cvt(c2w, c2w_bf, 64 * 512);   cvt(c3w, c3w_bf, 64 * 64);

  // Per-stage (row-field -> x5 flat) strides for fields (f3=b, f2, f1, f0); k stride = 4096 (c).
  // stage1: f2=d,f1=h,f0=w   stage2: f2=h,f1=d,f0=w   stage3: f2=w,f1=d,f0=h
  const long ST[3][4] = { {262144, 256, 16, 1}, {262144, 16, 256, 1}, {262144, 1, 256, 16} };
  // XW output strides, t-major: addr = (t*1024 + col)*512 + r  with GEMM row
  // M = r*16 + t  ->  r = f3*256 + f2*16 + f1, t = f0.
  const long XO[4] = {256, 16, 1, 524288};   // ok = 512
  // v_out read strides for horizontal XW rows m = n*256 + w*16 + h
  const long VR[4] = {2097152, 131072, 512, 8192};
  // h_out read strides for conv2 rows m = n*256 + h*16 + w
  const long HR[4] = {2097152, 131072, 512, 8192};

  const float* cur_in = x;
  float* outs[3] = {x5a, x5b, x5a};
  for (int s = 0; s < 3; ++s) {
    const long* S = ST[s];
    float* cur_out = outs[s];
    // vertical input projection, fwd + bwd  (M=8192, N=1024, K=64)
    gemm_bias_wmma<true, false><<<4096, 256, 0, stream>>>(
        cur_in, S[0], S[1], S[2], S[3], 4096, vwf_bf, 64, vbf,
        XWf, XO[0], XO[1], XO[2], XO[3], 512, 512, 64);
    gemm_bias_wmma<true, false><<<4096, 256, 0, stream>>>(
        cur_in, S[0], S[1], S[2], S[3], 4096, vwb_bf, 64, vbb,
        XWb, XO[0], XO[1], XO[2], XO[3], 512, 512, 64);
    // vertical biLSTM scan
    lstm_scan_wmma<<<dim3(32, 2), 256, 0, stream>>>(XWf, XWb, vuf_bf, vub_bf, vout);
    // horizontal input projection (M=8192, N=1024, K=512), A = permuted v_out (bf16)
    gemm_bias_wmma<false, false><<<4096, 256, 0, stream>>>(
        vout, VR[0], VR[1], VR[2], VR[3], 1, hwf_bf, 512, hbf,
        XWf, XO[0], XO[1], XO[2], XO[3], 512, 512, 64);
    gemm_bias_wmma<false, false><<<4096, 256, 0, stream>>>(
        vout, VR[0], VR[1], VR[2], VR[3], 1, hwb_bf, 512, hbb,
        XWb, XO[0], XO[1], XO[2], XO[3], 512, 512, 64);
    // horizontal biLSTM scan
    lstm_scan_wmma<<<dim3(32, 2), 256, 0, stream>>>(XWf, XWb, huf_bf, hub_bf, hout);
    // 1x1 conv (M=8192, N=64, K=512), write fp32 back into the stage's x5 positions
    gemm_bias_wmma<false, true><<<256, 256, 0, stream>>>(
        hout, HR[0], HR[1], HR[2], HR[3], 1, c2w_bf, 512, c2b,
        cur_out, S[0], S[1], S[2], S[3], 4096, 512, 4);
    cur_in = cur_out;
  }

  // Conv3d 1x1x1: (M=8192 rows = b,d,h,w; N=64, K=64), fp32 in -> fp32 out
  gemm_bias_wmma<true, true><<<256, 256, 0, stream>>>(
      x5a, 262144, 256, 16, 1, 4096, c3w_bf, 64, c3b,
      d_out, 262144, 256, 16, 1, 4096, 512, 4);
}
